// TemporalSynapticLearnableWeights_47648367182494
// MI455X (gfx1250) — matrix operand order repaired and checked by reading, not compiled, not run on hardware
//
#include <hip/hip_runtime.h>

// ---------------------------------------------------------------------------
// Problem constants (from reference)
// ---------------------------------------------------------------------------
constexpr int kB   = 4;
constexpr int kT   = 12;
constexpr int kN   = 2048;
constexpr int kFin = 26;
constexpr int kHid = 128;
constexpr int kLF  = 32;
constexpr int kFF  = 256;
constexpr int kHor = 12;
constexpr int kE   = 16384;
constexpr int kMR  = kB * kT * kN;        // 98304 flattened (b,t,n) rows
constexpr float kAlpha = 0.9f;
constexpr float kBeta  = 0.8f;
constexpr float kEps   = 1e-5f;

typedef __bf16  v16bf __attribute__((ext_vector_type(16)));
typedef float   v8f   __attribute__((ext_vector_type(8)));
typedef float   f32x4 __attribute__((ext_vector_type(4)));

// GEMM epilogue flags
constexpr int F_TRANSB = 1;   // B is stored [N,K] (use B^T)
constexpr int F_RELU   = 2;   // relu on output
constexpr int F_ACC    = 4;   // C += result (skip accumulator)

// LDS tile stride: 36 floats. Multiple of 4 -> every fragment gather and
// vector store is 16B aligned (ds_load_b128/ds_store_b128); 36 mod 64 banks
// still staggers the strided accesses.
constexpr int kLds = 36;

// ---------------------------------------------------------------------------
// Generic batched WMMA GEMM:
//   C[z] = act( A[z][M,K] @ op(B[z])[K,N] + bias + add0 + add1 + addmod ) (+C)
//
// Block = 128 threads (4 waves). Block tile = 64(M) x 64(N); each wave owns a
// 64x16 C strip: 4 x v_wmma_f32_16x16x32_bf16 per K-step sharing ONE B
// fragment. All five fragments (B + 4xA) are packed BEFORE the WMMA chain so
// the four WMMAs issue back-to-back (one co-exec hazard NOP batch per K-step
// instead of four).
//
// Staging: block-uniform scalar branch (no exec divergence) selects
//  - vector path (lda/ldb % 4 == 0, full tile): float4 global_load_b128 +
//    float4 LDS stores, fully unrolled & register-staged,
//  - scalar edge path: clamped address + select-zero b32 loads.
//
// Per-lane packing follows the CDNA5 ISA VGPR layouts:
//   A 16x32 bf16: lane l holds row (l&15); element j -> k = (j&7)+(j>=8)*16+(l>=16)*8
//   B 32x16 bf16: lane l holds col (l&15); element j -> k = (j&7)+(j>=8)*16+(l>=16)*8
//   C 16x16 f32 : lane l holds col (l&15); vgpr r -> row r + (l>=16)*8
// ---------------------------------------------------------------------------
__global__ __launch_bounds__(128) void gemm_wmma_kernel(
    const float* __restrict__ A, long long sA, int lda,
    const float* __restrict__ Bm, long long sB, int ldb,
    float* __restrict__ C, long long sC, int ldc,
    const float* __restrict__ bias,
    const float* __restrict__ add0, int ld0,
    const float* __restrict__ add1, int ld1,
    const float* __restrict__ addmod, int modN, int ldmod,
    int M, int N, int K, int flags)
{
  const int tid  = threadIdx.x;
  const int lane = tid & 31;
  const int wave = tid >> 5;
  const int hi   = lane >> 4;        // 0: lanes 0-15, 1: lanes 16-31
  const int lm   = lane & 15;
  const int m0   = blockIdx.y * 64;
  const int nblk = blockIdx.x * 64;
  const int z    = blockIdx.z;
  const bool trb = (flags & F_TRANSB) != 0;

  A  += (long long)z * sA;
  Bm += (long long)z * sB;
  C  += (long long)z * sC;

  __shared__ __align__(16) float As[64 * kLds];   // [row][k]
  __shared__ __align__(16) float Bs[64 * kLds];   // [n][k] (transposed tile)

  v8f acc[4] = {{}, {}, {}, {}};

  // Block-uniform fast-path tests (scalar branches, no exec divergence).
  const bool fullMN = (m0 + 64 <= M) && (nblk + 64 <= N);
  const bool vecOK  = ((lda & 3) == 0) && ((ldb & 3) == 0);

  for (int k0 = 0; k0 < K; k0 += 32) {
    const bool fast = fullMN && vecOK && (k0 + 32 <= K);

    if (fast) {
      // ---------- vector staging: float4 loads, register-staged ----------
      f32x4 va[4], vb[4];
#pragma unroll
      for (int u = 0; u < 4; ++u) {             // A: 64x32, coalesced in k
        int fi = tid + u * 128;                 // float4 index
        int r = fi >> 3, kc = (fi & 7) << 2;
        va[u] = *(const f32x4*)&A[(long long)(m0 + r) * lda + (k0 + kc)];
      }
      if (trb) {
#pragma unroll
        for (int u = 0; u < 4; ++u) {           // B^T: coalesced in k
          int fi = tid + u * 128;
          int n = fi >> 3, kc = (fi & 7) << 2;
          vb[u] = *(const f32x4*)&Bm[(long long)(nblk + n) * ldb + (k0 + kc)];
        }
#pragma unroll
        for (int u = 0; u < 4; ++u) {
          int fi = tid + u * 128;
          int n = fi >> 3, kc = (fi & 7) << 2;
          *(f32x4*)&Bs[n * kLds + kc] = vb[u];
        }
      } else {
#pragma unroll
        for (int u = 0; u < 4; ++u) {           // B: coalesced in n
          int fi = tid + u * 128;
          int kc = fi >> 4, n = (fi & 15) << 2;
          vb[u] = *(const f32x4*)&Bm[(long long)(k0 + kc) * ldb + (nblk + n)];
        }
#pragma unroll
        for (int u = 0; u < 4; ++u) {           // scatter into [n][k]
          int fi = tid + u * 128;
          int kc = fi >> 4, n = (fi & 15) << 2;
          Bs[(n + 0) * kLds + kc] = vb[u].x;
          Bs[(n + 1) * kLds + kc] = vb[u].y;
          Bs[(n + 2) * kLds + kc] = vb[u].z;
          Bs[(n + 3) * kLds + kc] = vb[u].w;
        }
      }
#pragma unroll
      for (int u = 0; u < 4; ++u) {
        int fi = tid + u * 128;
        int r = fi >> 3, kc = (fi & 7) << 2;
        *(f32x4*)&As[r * kLds + kc] = va[u];
      }
    } else {
      // ---------- edge tiles: clamped address + select-zero, scalar ------
      float va[16], vb[16];
#pragma unroll
      for (int u = 0; u < 16; ++u) {
        int i = tid + u * 128;
        int r = i >> 5, kc = i & 31;
        int gm = m0 + r, gk = k0 + kc;
        int cm = gm < M ? gm : M - 1;
        int ck = gk < K ? gk : K - 1;
        float v = A[(long long)cm * lda + ck];
        va[u] = (gm < M && gk < K) ? v : 0.0f;
      }
      if (trb) {
#pragma unroll
        for (int u = 0; u < 16; ++u) {
          int i = tid + u * 128;
          int n = i >> 5, kc = i & 31;
          int gn = nblk + n, gk = k0 + kc;
          int cn = gn < N ? gn : N - 1;
          int ck = gk < K ? gk : K - 1;
          float v = Bm[(long long)cn * ldb + ck];
          vb[u] = (gn < N && gk < K) ? v : 0.0f;
        }
#pragma unroll
        for (int u = 0; u < 16; ++u) {
          int i = tid + u * 128;
          Bs[(i >> 5) * kLds + (i & 31)] = vb[u];
        }
      } else {
#pragma unroll
        for (int u = 0; u < 16; ++u) {
          int i = tid + u * 128;
          int kc = i >> 6, n = i & 63;
          int gk = k0 + kc, gn = nblk + n;
          int cn = gn < N ? gn : N - 1;
          int ck = gk < K ? gk : K - 1;
          float v = Bm[(long long)ck * ldb + cn];
          vb[u] = (gk < K && gn < N) ? v : 0.0f;
        }
#pragma unroll
        for (int u = 0; u < 16; ++u) {
          int i = tid + u * 128;
          Bs[(i & 63) * kLds + (i >> 6)] = vb[u];
        }
      }
#pragma unroll
      for (int u = 0; u < 16; ++u) {
        int i = tid + u * 128;
        As[(i >> 5) * kLds + (i & 31)] = va[u];
      }
    }
    __syncthreads();

    // ---- pack ALL fragments first (b128 LDS gathers), then WMMA chain ----
    v16bf b, afr[4];
    {
      const f32x4* pb = (const f32x4*)&Bs[(wave * 16 + lm) * kLds + (hi << 3)];
#pragma unroll
      for (int q = 0; q < 4; ++q) {
        f32x4 t = pb[q + (q & 2)];
        b[4 * q + 0] = (__bf16)t.x;  b[4 * q + 1] = (__bf16)t.y;
        b[4 * q + 2] = (__bf16)t.z;  b[4 * q + 3] = (__bf16)t.w;
      }
    }
#pragma unroll
    for (int s4 = 0; s4 < 4; ++s4) {
      const f32x4* pa = (const f32x4*)&As[(s4 * 16 + lm) * kLds + (hi << 3)];
      const f32x4* qa = (const f32x4*)&As[(s4 * 16 + lm) * kLds + 16 + (hi << 3)];
#pragma unroll
      for (int q = 0; q < 2; ++q) {
        f32x4 t = pa[q];
        afr[s4][4 * q + 0] = (__bf16)t.x;  afr[s4][4 * q + 1] = (__bf16)t.y;
        afr[s4][4 * q + 2] = (__bf16)t.z;  afr[s4][4 * q + 3] = (__bf16)t.w;
        f32x4 u = qa[q];
        afr[s4][8 + 4 * q + 0] = (__bf16)u.x;  afr[s4][8 + 4 * q + 1] = (__bf16)u.y;
        afr[s4][8 + 4 * q + 2] = (__bf16)u.z;  afr[s4][8 + 4 * q + 3] = (__bf16)u.w;
      }
    }
#pragma unroll
    for (int s4 = 0; s4 < 4; ++s4) {
      acc[s4] = __builtin_amdgcn_wmma_f32_16x16x32_bf16(
          false, afr[s4], false, b, (short)0, acc[s4], false, false);
    }
    __syncthreads();
  }

  // ---- epilogue / store: four 16-row blocks per wave ----
  const int col = nblk + wave * 16 + lm;
  if (col < N) {
    const float bval = bias ? bias[col] : 0.0f;
#pragma unroll
    for (int s4 = 0; s4 < 4; ++s4) {
      const int rbase = m0 + s4 * 16 + (hi << 3);
#pragma unroll
      for (int r = 0; r < 8; ++r) {
        int row = rbase + r;
        if (row < M) {
          float v = acc[s4][r] + bval;
          if (add0)   v += add0[(long long)row * ld0 + col];
          if (add1)   v += add1[(long long)row * ld1 + col];
          if (addmod) v += addmod[(long long)(row % modN) * ldmod + col];
          long long ci = (long long)row * ldc + col;
          if (flags & F_ACC)  v += C[ci];
          if (flags & F_RELU) v = fmaxf(v, 0.0f);
          C[ci] = v;
        }
      }
    }
  }
}

// ---------------------------------------------------------------------------
// Elementwise / small kernels
// ---------------------------------------------------------------------------
__global__ void zero_kernel(float* p, long long n) {
  long long i = (long long)blockIdx.x * blockDim.x + threadIdx.x;
  if (i < n) p[i] = 0.0f;
}

__global__ void relu_kernel(const float* x, float* y, long long n) {
  long long i = (long long)blockIdx.x * blockDim.x + threadIdx.x;
  if (i < n) y[i] = fmaxf(x[i], 0.0f);
}

// h_out[row, 0:pc] = h[row, :]; h_out[row, pc:pc+LF] = lw[row % N, :]
__global__ void concat_lw_kernel(const float* __restrict__ h, int pc,
                                 const float* __restrict__ lw,
                                 float* __restrict__ out, int c, long long rows) {
  long long i = (long long)blockIdx.x * blockDim.x + threadIdx.x;
  long long total = rows * c;
  if (i >= total) return;
  int ch = (int)(i % c);
  long long row = i / c;
  int n = (int)(row % kN);
  out[i] = (ch < pc) ? h[row * pc + ch] : lw[(long long)n * kLF + (ch - pc)];
}

// 2nd-order LIF recurrence along T: syn=a*syn+x; mem=b*mem+syn; out=mem
__global__ void lif_scan_kernel(const float* __restrict__ in,
                                float* __restrict__ out, int c) {
  long long i = (long long)blockIdx.x * blockDim.x + threadIdx.x;
  long long total = (long long)kB * kN * c;
  if (i >= total) return;
  int ch = (int)(i % c);
  int n  = (int)((i / c) % kN);
  int b  = (int)(i / ((long long)c * kN));
  float syn = 0.0f, mem = 0.0f;
  for (int t = 0; t < kT; ++t) {
    long long off = (((long long)(b * kT + t) * kN + n) * c + ch);
    syn = kAlpha * syn + in[off];
    mem = kBeta * mem + syn;
    out[off] = mem;
  }
}

__global__ void deg_zero_kernel(float* degf, float* degb) {
  int i = blockIdx.x * blockDim.x + threadIdx.x;
  if (i < kN) { degf[i] = 0.0f; degb[i] = 0.0f; }
}

__global__ void deg_accum_kernel(const int* __restrict__ src,
                                 const int* __restrict__ dst,
                                 const float* __restrict__ ew,
                                 float* degf, float* degb) {
  int e = blockIdx.x * blockDim.x + threadIdx.x;
  if (e < kE) {
    atomicAdd(&degf[dst[e]], ew[e]);
    atomicAdd(&degb[src[e]], ew[e]);
  }
}

__global__ void edge_norm_kernel(const int* __restrict__ src,
                                 const int* __restrict__ dst,
                                 const float* __restrict__ ew,
                                 const float* __restrict__ degf,
                                 const float* __restrict__ degb,
                                 float* wf, float* wb) {
  int e = blockIdx.x * blockDim.x + threadIdx.x;
  if (e < kE) {
    wf[e] = ew[e] / fmaxf(degf[dst[e]], 1e-6f);
    wb[e] = ew[e] / fmaxf(degb[src[e]], 1e-6f);
  }
}

// Sparse diffusion step: zout[bt, dst, :] += w[e] * zin[bt, src, :]
// grid = (E, B*T); blockDim.x == c (160/192 -> 5/6 full waves).
__global__ void prop_kernel(const float* __restrict__ zin, int ldin,
                            float* __restrict__ zout, int ldout,
                            const int* __restrict__ src,
                            const int* __restrict__ dst,
                            const float* __restrict__ w, int c) {
  int e  = blockIdx.x;
  int bt = blockIdx.y;
  int ch = threadIdx.x;
  if (ch >= c) return;
  int s = src[e], d = dst[e];
  float v = zin[((long long)bt * kN + s) * ldin + ch] * w[e];
  atomicAdd(&zout[((long long)bt * kN + d) * ldout + ch], v);
}

// Per-channel mean / rsqrt(var+eps) over all rows. One block per channel.
__global__ void bn_stats_kernel(const float* __restrict__ x, int rows, int c,
                                float* __restrict__ stats) {
  int ch = blockIdx.x;
  float s = 0.0f, s2 = 0.0f;
  for (int r = threadIdx.x; r < rows; r += blockDim.x) {
    float v = x[(long long)r * c + ch];
    s += v; s2 += v * v;
  }
  __shared__ float sh[512];
  sh[threadIdx.x] = s;
  sh[256 + threadIdx.x] = s2;
  __syncthreads();
  for (int st = 128; st > 0; st >>= 1) {
    if (threadIdx.x < st) {
      sh[threadIdx.x] += sh[threadIdx.x + st];
      sh[256 + threadIdx.x] += sh[256 + threadIdx.x + st];
    }
    __syncthreads();
  }
  if (threadIdx.x == 0) {
    float mu = sh[0] / rows;
    float var = sh[256] / rows - mu * mu;
    stats[ch] = mu;
    stats[c + ch] = rsqrtf(var + kEps);
  }
}

__global__ void bn_apply_kernel(const float* __restrict__ x,
                                float* __restrict__ y,
                                const float* __restrict__ stats,
                                const float* __restrict__ g,
                                const float* __restrict__ b,
                                long long total, int c) {
  long long i = (long long)blockIdx.x * blockDim.x + threadIdx.x;
  if (i >= total) return;
  int ch = (int)(i % c);
  y[i] = (x[i] - stats[ch]) * stats[c + ch] * g[ch] + b[ch];
}

// Row softmax over a [kN, kN] matrix (post-ReLU values), in place.
__global__ void softmax_rows_kernel(float* __restrict__ a) {
  int row = blockIdx.x;
  float* p = a + (long long)row * kN;
  __shared__ float sh[256];
  float mx = -1e30f;
  for (int i = threadIdx.x; i < kN; i += 256) mx = fmaxf(mx, p[i]);
  sh[threadIdx.x] = mx;
  __syncthreads();
  for (int st = 128; st > 0; st >>= 1) {
    if (threadIdx.x < st) sh[threadIdx.x] = fmaxf(sh[threadIdx.x], sh[threadIdx.x + st]);
    __syncthreads();
  }
  mx = sh[0];
  __syncthreads();
  float s = 0.0f;
  for (int i = threadIdx.x; i < kN; i += 256) {
    float e = __expf(p[i] - mx);
    p[i] = e;
    s += e;
  }
  sh[threadIdx.x] = s;
  __syncthreads();
  for (int st = 128; st > 0; st >>= 1) {
    if (threadIdx.x < st) sh[threadIdx.x] += sh[threadIdx.x + st];
    __syncthreads();
  }
  float inv = 1.0f / sh[0];
  for (int i = threadIdx.x; i < kN; i += 256) p[i] *= inv;
}

// y[(b*N+n), hor*Fin+f] -> out[b, hor, n, f]
__global__ void write_out_kernel(const float* __restrict__ y,
                                 float* __restrict__ out) {
  long long idx = (long long)blockIdx.x * blockDim.x + threadIdx.x;
  long long total = (long long)kB * kHor * kN * kFin;
  if (idx >= total) return;
  int f = (int)(idx % kFin);
  long long t = idx / kFin;
  int n = (int)(t % kN); t /= kN;
  int hh = (int)(t % kHor);
  int b = (int)(t / kHor);
  out[idx] = y[((long long)b * kN + n) * (kHor * kFin) + hh * kFin + f];
}

// ---------------------------------------------------------------------------
// Host-side helpers
// ---------------------------------------------------------------------------
static inline void gemm(hipStream_t s,
                        const float* A, long long sA, int lda,
                        const float* B, long long sB, int ldb,
                        float* C, long long sC, int ldc,
                        const float* bias,
                        const float* add0, int ld0,
                        const float* add1, int ld1,
                        const float* addmod, int modN, int ldmod,
                        int M, int N, int K, int batch, int flags) {
  dim3 grid((N + 63) / 64, (M + 63) / 64, batch);
  gemm_wmma_kernel<<<grid, 128, 0, s>>>(A, sA, lda, B, sB, ldb, C, sC, ldc,
                                        bias, add0, ld0, add1, ld1,
                                        addmod, modN, ldmod, M, N, K, flags);
}

static inline void launch_zero(hipStream_t s, float* p, long long n) {
  zero_kernel<<<(unsigned)((n + 255) / 256), 256, 0, s>>>(p, n);
}

extern "C" void kernel_launch(void* const* d_in, const int* in_sizes, int n_in,
                              void* d_out, int out_size, void* d_ws, size_t ws_size,
                              hipStream_t stream) {
  (void)in_sizes; (void)n_in; (void)out_size; (void)ws_size;

  // ---- inputs (setup_inputs dict order) ----
  const float* x        = (const float*)d_in[0];
  const int*   eidx     = (const int*)d_in[1];   // [2, E]
  const float* ew       = (const float*)d_in[2];
  const float* enc_w    = (const float*)d_in[3];
  const float* enc_b    = (const float*)d_in[4];
  const float* node_emb = (const float*)d_in[5];
  const float* src_emb  = (const float*)d_in[6];
  const float* tgt_emb  = (const float*)d_in[7];
  struct Blk { const float *lw,*tW,*tb,*dcw,*dcb,*dsw,*dsb,*skw,*skb,*ng,*nb; };
  Blk blk[2];
  for (int i = 0; i < 2; ++i) {
    int o = 8 + i * 11;
    blk[i].lw  = (const float*)d_in[o + 0];
    blk[i].tW  = (const float*)d_in[o + 1];
    blk[i].tb  = (const float*)d_in[o + 2];
    blk[i].dcw = (const float*)d_in[o + 3];
    blk[i].dcb = (const float*)d_in[o + 4];
    blk[i].dsw = (const float*)d_in[o + 5];
    blk[i].dsb = (const float*)d_in[o + 6];
    blk[i].skw = (const float*)d_in[o + 7];
    blk[i].skb = (const float*)d_in[o + 8];
    blk[i].ng  = (const float*)d_in[o + 9];
    blk[i].nb  = (const float*)d_in[o + 10];
  }
  const float* rw1 = (const float*)d_in[30];
  const float* rb1 = (const float*)d_in[31];
  const float* rw2 = (const float*)d_in[32];
  const float* rb2 = (const float*)d_in[33];

  const int* srcI = eidx;           // edge_index[0]
  const int* dstI = eidx + kE;      // edge_index[1]

  // ---- workspace carve (float arena, 256B aligned) ----
  char* wptr = (char*)d_ws;
  auto carve = [&](long long elems) -> float* {
    float* p = (float*)wptr;
    wptr += ((elems * 4 + 255) / 256) * 256;
    return p;
  };
  const int CMAX = 192;
  float* hbuf  = carve((long long)kMR * CMAX);        // current hidden state
  float* resb  = carve((long long)kMR * CMAX);        // block residual (post concat)
  float* tmpb  = carve((long long)kMR * CMAX);        // gemm scratch
  float* zcat  = carve((long long)kMR * 4 * CMAX);    // diffusion outputs, 4 slices
  float* cat2  = carve((long long)kMR * 2 * CMAX);    // dense conv [x1|x2]
  float* xsb   = carve((long long)kMR * CMAX);        // diffconv mix output
  float* adjb  = carve((long long)kN * kN);           // learned adjacency
  float* skip  = carve((long long)kB * kN * kFF);     // skip accumulator (t = T-1 only)
  float* rbuf  = carve((long long)kB * kN * kFF);
  float* r2b   = carve((long long)kB * kN * 2 * kFF);
  float* ybuf  = carve((long long)kB * kN * kHor * kFin);
  float* wf    = carve(kE);
  float* wb    = carve(kE);
  float* degf  = carve(kN);
  float* degb  = carve(kN);
  float* stats = carve(2 * CMAX);

  // ---- edge transition weights (forward / backward random walks) ----
  deg_zero_kernel<<<(kN + 255) / 256, 256, 0, stream>>>(degf, degb);
  deg_accum_kernel<<<(kE + 255) / 256, 256, 0, stream>>>(srcI, dstI, ew, degf, degb);
  edge_norm_kernel<<<(kE + 255) / 256, 256, 0, stream>>>(srcI, dstI, ew, degf, degb, wf, wb);

  // ---- encoder: h = x @ enc_w + enc_b + node_emb[n]  ([MR,26]@[26,128]) ----
  gemm(stream, x, 0, kFin, enc_w, 0, kHid, hbuf, 0, kHid,
       enc_b, nullptr, 0, nullptr, 0, node_emb, kN, kHid,
       kMR, kHid, kFin, 1, 0);

  // ---- adjacency: softmax(relu(src_emb @ tgt_emb^T)) over rows ----
  gemm(stream, src_emb, 0, kHid, tgt_emb, 0, kHid, adjb, 0, kN,
       nullptr, nullptr, 0, nullptr, 0, nullptr, 1, 0,
       kN, kN, kHid, 1, F_TRANSB | F_RELU);
  softmax_rows_kernel<<<kN, 256, 0, stream>>>(adjb);

  // ---- skip accumulator (only t = T-1 is ever read by the readout) ----
  launch_zero(stream, skip, (long long)kB * kN * kFF);

  int pc = kHid;
  for (int i = 0; i < 2; ++i) {
    const int c = (i == 0) ? (kHid + kLF) : (kHid + 2 * kLF); // 160, 192
    const Blk& p = blk[i];
    const long long rows = kMR;

    // LearnableWeight concat -> residual buffer
    concat_lw_kernel<<<(unsigned)((rows * c + 255) / 256), 256, 0, stream>>>(
        hbuf, pc, p.lw, resb, c, rows);

    // Synaptic chain: 3x (Linear over features, then LIF scan over T)
    const float* cur = resb;
    for (int l = 0; l < 3; ++l) {
      gemm(stream, cur, 0, c, p.tW + (long long)l * c * c, 0, c, tmpb, 0, c,
           p.tb + (long long)l * c, nullptr, 0, nullptr, 0, nullptr, 1, 0,
           kMR, c, c, 1, 0);
      lif_scan_kernel<<<(unsigned)(((long long)kB * kN * c + 255) / 256), 256, 0, stream>>>(
          tmpb, hbuf, c);
      cur = hbuf;
    }

    // Skip: skip += h[:, T-1] @ skw + skb   (batched over B, M=kN each)
    gemm(stream, hbuf + (long long)(kT - 1) * kN * c, (long long)kT * kN * c, c,
         p.skw, 0, kFF, skip, (long long)kN * kFF, kFF,
         p.skb, nullptr, 0, nullptr, 0, nullptr, 1, 0,
         kN, kFF, c, kB, F_ACC);

    // Diffusion conv: 2 forward + 2 backward propagation orders into zcat slices
    launch_zero(stream, zcat, rows * 4 * c);
    {
      dim3 pg(kE, kB * kT);
      prop_kernel<<<pg, c, 0, stream>>>(hbuf, c,             zcat,         4 * c, srcI, dstI, wf, c);
      prop_kernel<<<pg, c, 0, stream>>>(zcat, 4 * c,         zcat + c,     4 * c, srcI, dstI, wf, c);
      prop_kernel<<<pg, c, 0, stream>>>(hbuf, c,             zcat + 2 * c, 4 * c, dstI, srcI, wb, c);
      prop_kernel<<<pg, c, 0, stream>>>(zcat + 2 * c, 4 * c, zcat + 3 * c, 4 * c, dstI, srcI, wb, c);
    }
    // xs = zcat @ dcw + dcb   ([MR,4c]@[4c,c])
    gemm(stream, zcat, 0, 4 * c, p.dcw, 0, c, xsb, 0, c,
         p.dcb, nullptr, 0, nullptr, 0, nullptr, 1, 0,
         kMR, c, 4 * c, 1, 0);

    // Dense graph conv: x1 = adj @ h[bt], x2 = adj @ x1  (48-way batched,
    // strideA = 0 keeps the 16.7MB adjacency resident in L2 across batches)
    gemm(stream, adjb, 0, kN, hbuf, (long long)kN * c, c,
         cat2, (long long)kN * 2 * c, 2 * c,
         nullptr, nullptr, 0, nullptr, 0, nullptr, 1, 0,
         kN, c, kN, kB * kT, 0);
    gemm(stream, adjb, 0, kN, cat2, (long long)kN * 2 * c, 2 * c,
         cat2 + c, (long long)kN * 2 * c, 2 * c,
         nullptr, nullptr, 0, nullptr, 0, nullptr, 1, 0,
         kN, c, kN, kB * kT, 0);
    // h = [x1|x2] @ dsw + dsb + xs + res   (fused epilogue)
    gemm(stream, cat2, 0, 2 * c, p.dsw, 0, c, tmpb, 0, c,
         p.dsb, xsb, c, resb, c, nullptr, 1, 0,
         kMR, c, 2 * c, 1, 0);

    // BatchNorm over (B,T,N) per channel
    bn_stats_kernel<<<c, 256, 0, stream>>>(tmpb, kMR, c, stats);
    bn_apply_kernel<<<(unsigned)((rows * c + 255) / 256), 256, 0, stream>>>(
        tmpb, hbuf, stats, p.ng, p.nb, rows * c, c);

    pc = c;
  }

  // ---- readout on last time step ----
  relu_kernel<<<(unsigned)(((long long)kB * kN * kFF + 255) / 256), 256, 0, stream>>>(
      skip, rbuf, (long long)kB * kN * kFF);
  gemm(stream, rbuf, 0, kFF, rw1, 0, 2 * kFF, r2b, 0, 2 * kFF,
       rb1, nullptr, 0, nullptr, 0, nullptr, 1, 0,
       kB * kN, 2 * kFF, kFF, 1, F_RELU);
  gemm(stream, r2b, 0, 2 * kFF, rw2, 0, kHor * kFin, ybuf, 0, kHor * kFin,
       rb2, nullptr, 0, nullptr, 0, nullptr, 1, 0,
       kB * kN, kHor * kFin, 2 * kFF, 1, 0);
  write_out_kernel<<<(unsigned)(((long long)kB * kHor * kN * kFin + 255) / 256), 256, 0, stream>>>(
      ybuf, (float*)d_out);
}
